// Router_74526272520644
// MI455X (gfx1250) — compile-verified, matrix operand order checked by reading, not run on hardware
//
#include <hip/hip_runtime.h>
#include <hip/hip_bf16.h>
#include <math.h>

// ---------------------------------------------------------------------------
// Problem constants (from the reference)
// ---------------------------------------------------------------------------
static constexpr int B_      = 2;
static constexpr int S_      = 512;
static constexpr int BS      = B_ * S_;      // 1024 tokens
static constexpr int D_      = 1024;
static constexpr int N_QK    = 2048;
static constexpr int N_V     = 2048;
static constexpr int N_KNOW  = 8192;
static constexpr int NC_QK   = 48;           // MAX_K_QK * CAND_MULT
static constexpr int NC_V    = 48;
static constexpr int NC_KNOW = 96;
static constexpr int K_QK    = 16;
static constexpr int K_V     = 16;
static constexpr int K_KNOW  = 32;
static constexpr float DROP_SCALE = 1.0f / 0.9f;   // 1/(1-DROP)

typedef __attribute__((ext_vector_type(16))) _Float16 v16h;
typedef __attribute__((ext_vector_type(8)))  _Float16 v8h;
typedef __attribute__((ext_vector_type(8)))  float    v8f;

// ---------------------------------------------------------------------------
// K0: zero the two loss accumulators (d_out is poisoned by the harness)
// ---------------------------------------------------------------------------
__global__ void zero_losses_kernel(float* a, float* b) {
    if (threadIdx.x == 0) { *a = 0.0f; *b = 0.0f; }
}

// ---------------------------------------------------------------------------
// K1: f32 -> f16 conversion with optional scale
// ---------------------------------------------------------------------------
__global__ void __launch_bounds__(256)
cvt_f16_kernel(const float* __restrict__ in, _Float16* __restrict__ out,
               float scale, int n) {
    int i = blockIdx.x * 256 + threadIdx.x;
    if (i < n) out[i] = (_Float16)(in[i] * scale);
}

// ---------------------------------------------------------------------------
// K2: per-token small projections: pos_qk/pos_v/pos_know (D->2), tau (D->3),
//     tau_know (D->1). One 256-thread block per token, LDS tree reduction.
// ---------------------------------------------------------------------------
__global__ void __launch_bounds__(256)
proj_kernel(const float* __restrict__ x,
            const float* __restrict__ Wqk, const float* __restrict__ bqk,
            const float* __restrict__ Wv,  const float* __restrict__ bv,
            const float* __restrict__ Wkn, const float* __restrict__ bkn,
            const float* __restrict__ Wt,  const float* __restrict__ bt,
            const float* __restrict__ Wtk, const float* __restrict__ btk,
            float* __restrict__ qk_pos, float* __restrict__ v_pos,
            float* __restrict__ know_pos, float* __restrict__ tau,
            float* __restrict__ tauk) {
    const int t   = blockIdx.x;
    const int tid = threadIdx.x;
    const float* xr = x + (size_t)t * D_;
    float acc[10];
#pragma unroll
    for (int j = 0; j < 10; ++j) acc[j] = 0.0f;
    for (int d = tid; d < D_; d += 256) {
        float xv = xr[d];
        acc[0] += xv * Wqk[d * 2 + 0];
        acc[1] += xv * Wqk[d * 2 + 1];
        acc[2] += xv * Wv [d * 2 + 0];
        acc[3] += xv * Wv [d * 2 + 1];
        acc[4] += xv * Wkn[d * 2 + 0];
        acc[5] += xv * Wkn[d * 2 + 1];
        acc[6] += xv * Wt [d * 3 + 0];
        acc[7] += xv * Wt [d * 3 + 1];
        acc[8] += xv * Wt [d * 3 + 2];
        acc[9] += xv * Wtk[d];
    }
    __shared__ float red[256];
    float out[10];
    for (int j = 0; j < 10; ++j) {
        red[tid] = acc[j];
        __syncthreads();
        for (int s = 128; s > 0; s >>= 1) {
            if (tid < s) red[tid] += red[tid + s];
            __syncthreads();
        }
        if (tid == 0) out[j] = red[0];
        __syncthreads();
    }
    if (tid == 0) {
        qk_pos  [t * 2 + 0] = out[0] + bqk[0];
        qk_pos  [t * 2 + 1] = out[1] + bqk[1];
        v_pos   [t * 2 + 0] = out[2] + bv[0];
        v_pos   [t * 2 + 1] = out[3] + bv[1];
        know_pos[t * 2 + 0] = out[4] + bkn[0];
        know_pos[t * 2 + 1] = out[5] + bkn[1];
        tau     [t * 3 + 0] = out[6] + bt[0];
        tau     [t * 3 + 1] = out[7] + bt[1];
        tau     [t * 3 + 2] = out[8] + bt[2];
        tauk    [t]         = out[9] + btk[0];
    }
}

// ---------------------------------------------------------------------------
// K3: squared-distance + top-ncand selection (ascending dist, ties -> lower
//     index, matching jax.lax.top_k(-dist)). One block per token; dist array
//     lives in dynamic LDS (<= 34KB of the WGP's 320KB).
// ---------------------------------------------------------------------------
__global__ void __launch_bounds__(256)
dist_topk_kernel(const float* __restrict__ tok_pos,   // [BS,2]
                 const float* __restrict__ npos,      // [N,2]
                 int N, int ncand,
                 int*   __restrict__ cand_idx,        // [BS,ncand]
                 float* __restrict__ cand_dist,       // [BS,ncand]
                 float* __restrict__ idx_out_f) {     // [BS,ncand] (d_out)
    extern __shared__ unsigned char smem_raw[];
    float* dist = (float*)smem_raw;            // N floats
    float* rv   = dist + N;                    // 256 floats
    int*   ri   = (int*)(rv + 256);            // 256 ints

    const int t   = blockIdx.x;
    const int tid = threadIdx.x;
    const float px = tok_pos[t * 2 + 0];
    const float py = tok_pos[t * 2 + 1];

    for (int i = tid; i < N; i += 256) {
        float dx = px - npos[i * 2 + 0];
        float dy = py - npos[i * 2 + 1];
        dist[i] = dx * dx + dy * dy;
    }
    __syncthreads();

    const float REMOVED = 3.402823466e38f;
    for (int c = 0; c < ncand; ++c) {
        float best = REMOVED;
        int   bi   = 0x7fffffff;
        for (int i = tid; i < N; i += 256) {
            float v = dist[i];
            if (v < best || (v == best && i < bi)) { best = v; bi = i; }
        }
        rv[tid] = best; ri[tid] = bi;
        __syncthreads();
        for (int s = 128; s > 0; s >>= 1) {
            if (tid < s) {
                float v = rv[tid + s]; int j = ri[tid + s];
                if (v < rv[tid] || (v == rv[tid] && j < ri[tid])) {
                    rv[tid] = v; ri[tid] = j;
                }
            }
            __syncthreads();
        }
        if (tid == 0) {
            int sel = ri[0];
            cand_idx [t * ncand + c] = sel;
            cand_dist[t * ncand + c] = rv[0];
            idx_out_f[t * ncand + c] = (float)sel;
            dist[sel] = REMOVED;
        }
        __syncthreads();
    }
}

// ---------------------------------------------------------------------------
// K4: dense score GEMM on the matrix pipes:  C[m,n] = sum_d X[m,d]*W[n,d]
//     X:[BS,D] f16 (drop-scaled), W:[N,D] f16, C:[BS,N] f32.
//     Register-blocked: one wave -> 16x64 of C (4 independent accumulators,
//     4 back-to-back v_wmma per k-step, A fragment reused 4x). 4 waves/block
//     -> 16x256 of C per block.
//     Lane layout per CDNA5 ISA 16-bit A/B maps:
//       lane<16 : row (lane&15), K = kb+0..7  and kb+16..23
//       lane>=16: row (lane&15), K = kb+8..15 and kb+24..31
// ---------------------------------------------------------------------------
__global__ void __launch_bounds__(128)
wmma_scores_kernel(const _Float16* __restrict__ X,
                   const _Float16* __restrict__ W,
                   float* __restrict__ C,
                   int N, int D) {
    const int lane   = threadIdx.x & 31;
    const int wave   = threadIdx.x >> 5;
    const int m0     = blockIdx.y * 16;
    const int n0     = blockIdx.x * 256 + wave * 64;   // 4 N-tiles per wave
    const int half16 = lane >> 4;        // 0 or 1
    const int r      = lane & 15;

    const _Float16* arow = X + (size_t)(m0 + r) * D + half16 * 8;
    const _Float16* brow = W + (size_t)(n0 + r) * D + half16 * 8;
    const size_t    bstp = (size_t)16 * D;             // next N-tile row block

    v8f acc0, acc1, acc2, acc3;
#pragma unroll
    for (int i = 0; i < 8; ++i) { acc0[i] = 0.0f; acc1[i] = 0.0f;
                                  acc2[i] = 0.0f; acc3[i] = 0.0f; }

    for (int k = 0; k < D; k += 32) {
        if (k + 32 < D) {
            __builtin_prefetch(arow + k + 32, 0, 1);           // global_prefetch_b8
            __builtin_prefetch(brow + k + 32, 0, 1);
            __builtin_prefetch(brow + bstp + k + 32, 0, 1);
            __builtin_prefetch(brow + 2 * bstp + k + 32, 0, 1);
            __builtin_prefetch(brow + 3 * bstp + k + 32, 0, 1);
        }
        v8h a0 = *(const v8h*)(arow + k);
        v8h a1 = *(const v8h*)(arow + k + 16);
        v16h a = __builtin_shufflevector(a0, a1, 0, 1, 2, 3, 4, 5, 6, 7,
                                                 8, 9, 10, 11, 12, 13, 14, 15);
        v16h b[4];
#pragma unroll
        for (int j = 0; j < 4; ++j) {
            const _Float16* bp = brow + (size_t)j * bstp + k;
            v8h b0 = *(const v8h*)(bp);
            v8h b1 = *(const v8h*)(bp + 16);
            b[j] = __builtin_shufflevector(b0, b1, 0, 1, 2, 3, 4, 5, 6, 7,
                                                   8, 9, 10, 11, 12, 13, 14, 15);
        }
        // 4 independent WMMAs: no D->A/B RAW hazard, dense XDL issue
        acc0 = __builtin_amdgcn_wmma_f32_16x16x32_f16(false, a, false, b[0],
                                                      (short)0, acc0, false, false);
        acc1 = __builtin_amdgcn_wmma_f32_16x16x32_f16(false, a, false, b[1],
                                                      (short)0, acc1, false, false);
        acc2 = __builtin_amdgcn_wmma_f32_16x16x32_f16(false, a, false, b[2],
                                                      (short)0, acc2, false, false);
        acc3 = __builtin_amdgcn_wmma_f32_16x16x32_f16(false, a, false, b[3],
                                                      (short)0, acc3, false, false);
    }

    // C/D layout: VGPR i -> M = m0 + half16*8 + i, N = ntile + (lane&15)
    const int mbase = m0 + half16 * 8;
    {
        float* cp = C + (size_t)mbase * N + (n0 + r);
#pragma unroll
        for (int i = 0; i < 8; ++i) cp[(size_t)i * N] = acc0[i];
    }
    {
        float* cp = C + (size_t)mbase * N + (n0 + 16 + r);
#pragma unroll
        for (int i = 0; i < 8; ++i) cp[(size_t)i * N] = acc1[i];
    }
    {
        float* cp = C + (size_t)mbase * N + (n0 + 32 + r);
#pragma unroll
        for (int i = 0; i < 8; ++i) cp[(size_t)i * N] = acc2[i];
    }
    {
        float* cp = C + (size_t)mbase * N + (n0 + 48 + r);
#pragma unroll
        for (int i = 0; i < 8; ++i) cp[(size_t)i * N] = acc3[i];
    }
}

// ---------------------------------------------------------------------------
// K5: gather candidate scores, threshold gate (exact kth-largest threshold),
//     write gates, accumulate mean(gate * dist) losses.
// ---------------------------------------------------------------------------
__global__ void __launch_bounds__(128)
gate_kernel(const float* __restrict__ scores, int N,
            const int*   __restrict__ cand_idx, int ncand, int maxk,
            const float* __restrict__ tau, int tau_stride, int tau_col,
            const float* __restrict__ cand_dist,
            float* __restrict__ gate_out,
            float* __restrict__ loss_accum, float loss_scale) {
    int t = blockIdx.x * 128 + threadIdx.x;
    if (t >= BS) return;

    float eg[96];
    float tmp[96];
    const float tv = tau[t * tau_stride + tau_col];
    float egmax = 0.0f;
    for (int i = 0; i < ncand; ++i) {
        int   idx = cand_idx[t * ncand + i];
        float s   = scores[(size_t)t * N + idx];
        float raw = s - tv;
        float g   = (raw > 0.0f) ? raw : 1e-8f * expf(raw);
        float e   = expf(g) - 1.0f;
        eg[i] = e; tmp[i] = e;
        if (e > egmax) egmax = e;
    }
    // kth-largest value (threshold of jax.lax.top_k)
    float thresh = egmax;
    for (int k = 0; k < maxk; ++k) {
        float m = -3.0e38f; int mi = 0;
        for (int i = 0; i < ncand; ++i) {
            if (tmp[i] > m) { m = tmp[i]; mi = i; }
        }
        thresh = m;
        tmp[mi] = -3.0e38f;
    }
    float gsum = 0.0f;
    for (int i = 0; i < ncand; ++i) {
        if (eg[i] >= thresh) gsum += eg[i];
        else                 eg[i] = 0.0f;
    }
    gsum += 1e-8f;
    const float strength = tanhf(egmax);
    float lpart = 0.0f;
    for (int i = 0; i < ncand; ++i) {
        float g = eg[i] / gsum * strength;
        gate_out[t * ncand + i] = g;
        if (loss_accum) lpart += g * cand_dist[t * ncand + i];
    }
    if (loss_accum) atomicAdd(loss_accum, lpart * loss_scale);
}

// ---------------------------------------------------------------------------
// Host-side orchestration
// ---------------------------------------------------------------------------
extern "C" void kernel_launch(void* const* d_in, const int* in_sizes, int n_in,
                              void* d_out, int out_size, void* d_ws, size_t ws_size,
                              hipStream_t stream) {
    (void)in_sizes; (void)n_in; (void)out_size; (void)ws_size;

    const float* x        = (const float*)d_in[0];
    const float* npos     = (const float*)d_in[1];   // [N_QK+N_V+N_KNOW, 2]
    const float* qk_neu   = (const float*)d_in[2];
    const float* v_neu    = (const float*)d_in[3];
    const float* kn_neu   = (const float*)d_in[4];
    const float* Wqk      = (const float*)d_in[5];
    const float* bqk      = (const float*)d_in[6];
    const float* Wv       = (const float*)d_in[7];
    const float* bv       = (const float*)d_in[8];
    const float* Wkn      = (const float*)d_in[9];
    const float* bkn      = (const float*)d_in[10];
    const float* Wt       = (const float*)d_in[11];
    const float* bt       = (const float*)d_in[12];
    const float* Wtk      = (const float*)d_in[13];
    const float* btk      = (const float*)d_in[14];

    // Workspace carve-up (256B-aligned segments)
    char* base = (char*)d_ws;
    size_t off = 0;
    auto alloc = [&](size_t bytes) -> void* {
        void* p = base + off;
        off += (bytes + 255) & ~(size_t)255;
        return p;
    };
    _Float16* xh   = (_Float16*)alloc((size_t)BS * D_ * 2);
    _Float16* qkh  = (_Float16*)alloc((size_t)N_QK * D_ * 2);
    _Float16* vh   = (_Float16*)alloc((size_t)N_V * D_ * 2);
    _Float16* knh  = (_Float16*)alloc((size_t)N_KNOW * D_ * 2);
    float* s_qk    = (float*)alloc((size_t)BS * N_QK * 4);
    float* s_v     = (float*)alloc((size_t)BS * N_V * 4);
    float* s_kn    = (float*)alloc((size_t)BS * N_KNOW * 4);
    float* p_qk    = (float*)alloc((size_t)BS * 2 * 4);
    float* p_v     = (float*)alloc((size_t)BS * 2 * 4);
    float* p_kn    = (float*)alloc((size_t)BS * 2 * 4);
    float* tau     = (float*)alloc((size_t)BS * 3 * 4);
    float* tauk    = (float*)alloc((size_t)BS * 4);
    int*   ci_qk   = (int*)alloc((size_t)BS * NC_QK * 4);
    int*   ci_v    = (int*)alloc((size_t)BS * NC_V * 4);
    int*   ci_kn   = (int*)alloc((size_t)BS * NC_KNOW * 4);
    float* cd_qk   = (float*)alloc((size_t)BS * NC_QK * 4);
    float* cd_v    = (float*)alloc((size_t)BS * NC_V * 4);
    float* cd_kn   = (float*)alloc((size_t)BS * NC_KNOW * 4);

    // Output layout (floats, reference tuple order, flattened)
    float* out      = (float*)d_out;
    float* o_gateQ  = out;                                    // BS*48
    float* o_gateK  = o_gateQ + (size_t)BS * NC_QK;           // BS*48
    float* o_gateV  = o_gateK + (size_t)BS * NC_QK;           // BS*48
    float* o_idxQK  = o_gateV + (size_t)BS * NC_V;            // BS*48
    float* o_idxV   = o_idxQK + (size_t)BS * NC_QK;           // BS*48
    float* o_lossA  = o_idxV  + (size_t)BS * NC_V;            // 1
    float* o_gateKn = o_lossA + 1;                            // BS*96
    float* o_idxKn  = o_gateKn + (size_t)BS * NC_KNOW;        // BS*96
    float* o_lossK  = o_idxKn  + (size_t)BS * NC_KNOW;        // 1

    // K0: zero loss accumulators
    zero_losses_kernel<<<1, 32, 0, stream>>>(o_lossA, o_lossK);

    // K1: f32 -> f16 (x gets the dropout scale baked in)
    {
        int n;
        n = BS * D_;
        cvt_f16_kernel<<<(n + 255) / 256, 256, 0, stream>>>(x, xh, DROP_SCALE, n);
        n = N_QK * D_;
        cvt_f16_kernel<<<(n + 255) / 256, 256, 0, stream>>>(qk_neu, qkh, 1.0f, n);
        n = N_V * D_;
        cvt_f16_kernel<<<(n + 255) / 256, 256, 0, stream>>>(v_neu, vh, 1.0f, n);
        n = N_KNOW * D_;
        cvt_f16_kernel<<<(n + 255) / 256, 256, 0, stream>>>(kn_neu, knh, 1.0f, n);
    }

    // K2: projections
    proj_kernel<<<BS, 256, 0, stream>>>(x, Wqk, bqk, Wv, bv, Wkn, bkn,
                                        Wt, bt, Wtk, btk,
                                        p_qk, p_v, p_kn, tau, tauk);

    // K3: distance + top-k candidate selection
    {
        size_t smem_qk = (size_t)(N_QK + 256) * 4 + 256 * 4;
        size_t smem_kn = (size_t)(N_KNOW + 256) * 4 + 256 * 4;
        dist_topk_kernel<<<BS, 256, smem_qk, stream>>>(
            p_qk, npos, N_QK, NC_QK, ci_qk, cd_qk, o_idxQK);
        dist_topk_kernel<<<BS, 256, smem_qk, stream>>>(
            p_v, npos + (size_t)N_QK * 2, N_V, NC_V, ci_v, cd_v, o_idxV);
        dist_topk_kernel<<<BS, 256, smem_kn, stream>>>(
            p_kn, npos + (size_t)(N_QK + N_V) * 2, N_KNOW, NC_KNOW,
            ci_kn, cd_kn, o_idxKn);
    }

    // K4: dense WMMA score GEMMs (each block: 16 rows x 256 cols of C)
    {
        dim3 blk(128, 1, 1);
        dim3 g_qk(N_QK / 256, BS / 16, 1);
        dim3 g_kn(N_KNOW / 256, BS / 16, 1);
        wmma_scores_kernel<<<g_qk, blk, 0, stream>>>(xh, qkh, s_qk, N_QK, D_);
        wmma_scores_kernel<<<g_qk, blk, 0, stream>>>(xh, vh,  s_v,  N_V,  D_);
        wmma_scores_kernel<<<g_kn, blk, 0, stream>>>(xh, knh, s_kn, N_KNOW, D_);
    }

    // K5: gating + losses
    {
        const int gblk = 128;
        const int ggrd = (BS + gblk - 1) / gblk;
        // gate_Q: tau col 0, contributes mean(gQ * dist_qk) to pos_loss_attn
        gate_kernel<<<ggrd, gblk, 0, stream>>>(
            s_qk, N_QK, ci_qk, NC_QK, K_QK, tau, 3, 0, cd_qk,
            o_gateQ, o_lossA, 1.0f / (float)(BS * NC_QK));
        // gate_K: same scores, tau col 1, no loss contribution
        gate_kernel<<<ggrd, gblk, 0, stream>>>(
            s_qk, N_QK, ci_qk, NC_QK, K_QK, tau, 3, 1, cd_qk,
            o_gateK, nullptr, 0.0f);
        // gate_V: tau col 2, contributes mean(gV * dist_v) to pos_loss_attn
        gate_kernel<<<ggrd, gblk, 0, stream>>>(
            s_v, N_V, ci_v, NC_V, K_V, tau, 3, 2, cd_v,
            o_gateV, o_lossA, 1.0f / (float)(BS * NC_V));
        // gate_know: tau_k, contributes mean(gKn * dist) to pos_loss_know
        gate_kernel<<<ggrd, gblk, 0, stream>>>(
            s_kn, N_KNOW, ci_kn, NC_KNOW, K_KNOW, tauk, 1, 0, cd_kn,
            o_gateKn, o_lossK, 1.0f / (float)(BS * NC_KNOW));
    }
}
